// MultiheadAttention_75720273429233
// MI455X (gfx1250) — matrix operand order repaired and checked by reading, not compiled, not run on hardware
//
#include <hip/hip_runtime.h>

// ---------------------------------------------------------------------------
// MI455X (gfx1250) multi-head attention pipeline, bf16 WMMA GEMM core.
// BM = 256 = M so each fp32 weight byte is read from HBM exactly once
// (weights are the 1.5 GB roofline term @ 23.3 TB/s); weight stream uses
// non-temporal loads so the 8 MB activation panels stay resident in the
// 192 MB L2. Double-buffered LDS staging, packed bf16 via v_perm_b32,
// fragments loaded as aligned ds_load_b128 pairs.
// ---------------------------------------------------------------------------

typedef __bf16 bf16_t;
typedef __attribute__((ext_vector_type(16))) __bf16        v16bf;
typedef __attribute__((ext_vector_type(8)))  float         v8f;
typedef __attribute__((ext_vector_type(4)))  unsigned int  v4u;
typedef __attribute__((ext_vector_type(2)))  unsigned int  v2u;

struct QuadPair { v4u lo, hi; };   // 32 B == v16bf

#define THREADS 256
#define BM 256            // == M for every GEMM in this problem
#define BN 64
#define BK 32
#define AS_STRIDE 40      // bf16 elems; row pitch 80 B (16B-aligned quads)
#define BS_STRIDE 40      // Bs stored transposed: [BN][BK]

// fp32 pair -> packed bf16 dword: round-to-nearest (ties away) + v_perm_b32
__device__ __forceinline__ unsigned int pack_bf2(float a, float b) {
  const unsigned int ua = __builtin_bit_cast(unsigned int, a) + 0x8000u;
  const unsigned int ub = __builtin_bit_cast(unsigned int, b) + 0x8000u;
  // result bytes: {ub.b3, ub.b2, ua.b3, ua.b2}  (sel 0-3 -> S1, 4-7 -> S0)
  return __builtin_amdgcn_perm(ub, ua, 0x07060302u);
}

// C[h] = A[h] (256xK fp32) @ B[h] (KxN fp32, or NxK if TRANSB) + bias -> fp32.
// gridDim.x = N/BN, gridDim.z = heads. K % BK == 0. M == 256 always.
template <bool TRANSB>
__global__ __launch_bounds__(THREADS)
void gemm_bf16_wmma(const float* __restrict__ A, int lda, size_t strideA,
                    const float* __restrict__ B, int ldb, size_t strideB,
                    const float* __restrict__ bias,
                    float* __restrict__ C, int ldc, size_t strideC,
                    int K)
{
  __shared__ __align__(16) bf16_t As[2][BM * AS_STRIDE];  // 2 x 20 KB
  __shared__ __align__(16) bf16_t Bs[2][BN * BS_STRIDE];  // 2 x  5 KB

  const int h = blockIdx.z;
  A += strideA * h;
  B += strideB * h;
  C += strideC * h;

  const int n0 = blockIdx.x * BN;

  const int tid   = threadIdx.x;
  const int lane  = tid & 31;
  const int wave  = tid >> 5;
  const int wm    = wave >> 1;   // 0..3  (M direction, 64 rows each)
  const int wn    = wave & 1;    // 0..1  (N direction, 32 cols each)
  const int l15   = lane & 15;
  const int lhalf = lane >> 4;

  v8f acc[4][2] = {};

  // ---- staged (packed bf16) tile data held in registers ------------------
  unsigned int a_pk[16];   // 8 float4 -> 16 dwords
  unsigned int b_pk[4];    // 8 floats -> 4 dwords

  // load global tile k0 into registers (packed)
  auto load_tiles = [&](int k0) {
    // A: 256x32 fp32, 8 float4 per thread, coalesced along k (L2-resident)
    #pragma unroll
    for (int i = 0; i < 8; ++i) {
      const int idx4 = tid + i * THREADS;          // 0..2047
      const int r = idx4 >> 3, c4 = idx4 & 7;
      const float4 f = *(const float4*)&A[(size_t)r * lda + k0 + 4 * c4];
      a_pk[2 * i + 0] = pack_bf2(f.x, f.y);
      a_pk[2 * i + 1] = pack_bf2(f.z, f.w);
    }
    if (!TRANSB) {
      // B row-major [K x N] weight stream: read-once -> non-temporal loads
      #pragma unroll
      for (int i = 0; i < 4; ++i) {
        const int idx = tid + i * THREADS;         // 0..1023
        const int n = idx & (BN - 1), kk2 = idx >> 6;   // kk2: 0..15
        const float f0 = __builtin_nontemporal_load(&B[(size_t)(k0 + 2 * kk2 + 0) * ldb + n0 + n]);
        const float f1 = __builtin_nontemporal_load(&B[(size_t)(k0 + 2 * kk2 + 1) * ldb + n0 + n]);
        b_pk[i] = pack_bf2(f0, f1);
      }
      // prefetch next weight tile (HBM stream) -> global_prefetch_b8
      if (k0 + BK < K)
        __builtin_prefetch(&B[(size_t)(k0 + BK + (tid >> 3)) * ldb + n0 + ((tid & 7) << 3)], 0, 1);
    } else {
      // B is [N x K] (Qq^T): float4 along k, 2 per thread
      #pragma unroll
      for (int i = 0; i < 2; ++i) {
        const int idx4 = tid + i * THREADS;        // 0..511
        const int n = idx4 >> 3, c4 = idx4 & 7;
        const float4 f = *(const float4*)&B[(size_t)(n0 + n) * ldb + k0 + 4 * c4];
        b_pk[2 * i + 0] = pack_bf2(f.x, f.y);
        b_pk[2 * i + 1] = pack_bf2(f.z, f.w);
      }
    }
  };

  // write packed registers into LDS buffer `buf`
  auto store_tiles = [&](int buf) {
    #pragma unroll
    for (int i = 0; i < 8; ++i) {
      const int idx4 = tid + i * THREADS;
      const int r = idx4 >> 3, c4 = idx4 & 7;
      v2u p; p[0] = a_pk[2 * i]; p[1] = a_pk[2 * i + 1];
      *(v2u*)&As[buf][r * AS_STRIDE + 4 * c4] = p;
    }
    if (!TRANSB) {
      #pragma unroll
      for (int i = 0; i < 4; ++i) {
        const int idx = tid + i * THREADS;
        const int n = idx & (BN - 1), kk2 = idx >> 6;
        *(unsigned int*)&Bs[buf][n * BS_STRIDE + 2 * kk2] = b_pk[i];
      }
    } else {
      #pragma unroll
      for (int i = 0; i < 2; ++i) {
        const int idx4 = tid + i * THREADS;
        const int n = idx4 >> 3, c4 = idx4 & 7;
        v2u p; p[0] = b_pk[2 * i]; p[1] = b_pk[2 * i + 1];
        *(v2u*)&Bs[buf][n * BS_STRIDE + 4 * c4] = p;
      }
    }
  };

  // 8 WMMAs on buffer `buf`; fragments as aligned ds_load_b128 pairs
  auto compute = [&](int buf) {
    const unsigned int* Asu = (const unsigned int*)&As[buf][0];
    const unsigned int* Bsu = (const unsigned int*)&Bs[buf][0];
    v16bf afrag[4];
    #pragma unroll
    for (int mf = 0; mf < 4; ++mf) {
      const int m = wm * 64 + mf * 16 + l15;
      // quads at byte offsets 80*m + 16*lhalf and 80*m + 32 + 16*lhalf
      QuadPair q;
      q.lo = *(const v4u*)(Asu + m * (AS_STRIDE / 2) + 4 * lhalf);
      q.hi = *(const v4u*)(Asu + m * (AS_STRIDE / 2) + 8 + 4 * lhalf);
      afrag[mf] = __builtin_bit_cast(v16bf, q);
    }
    v16bf bfrag[2];
    #pragma unroll
    for (int nf = 0; nf < 2; ++nf) {
      const int n = wn * 32 + nf * 16 + l15;
      QuadPair q;
      q.lo = *(const v4u*)(Bsu + n * (BS_STRIDE / 2) + 4 * lhalf);
      q.hi = *(const v4u*)(Bsu + n * (BS_STRIDE / 2) + 8 + 4 * lhalf);
      bfrag[nf] = __builtin_bit_cast(v16bf, q);
    }
    #pragma unroll
    for (int mf = 0; mf < 4; ++mf)
      #pragma unroll
      for (int nf = 0; nf < 2; ++nf)
        acc[mf][nf] = __builtin_amdgcn_wmma_f32_16x16x32_bf16(
            false, afrag[mf], false, bfrag[nf],
            (short)0, acc[mf][nf], false, false);
  };

  // ---- double-buffered main loop ----------------------------------------
  load_tiles(0);
  store_tiles(0);
  int cur = 0;
  for (int k0 = BK; k0 < K; k0 += BK) {
    load_tiles(k0);        // global loads for next tile in flight
    __syncthreads();       // buf[cur] fully written by all threads
    compute(cur);          // WMMAs overlap outstanding global loads
    store_tiles(cur ^ 1);
    cur ^= 1;
  }
  __syncthreads();
  compute(cur);

  // ---- epilogue: C layout (lane -> n = l15, VGPR r -> m = r + 8*lhalf) ---
  #pragma unroll
  for (int nf = 0; nf < 2; ++nf) {
    const int ng = n0 + wn * 32 + nf * 16 + l15;
    const float bv = bias ? bias[ng] : 0.0f;
    #pragma unroll
    for (int mf = 0; mf < 4; ++mf) {
      const int mbase = wm * 64 + mf * 16 + 8 * lhalf;
      #pragma unroll
      for (int r = 0; r < 8; ++r)
        C[(size_t)(mbase + r) * ldc + ng] = acc[mf][nf][r] + bv;
    }
  }
}

// out[h,n] = sum_m softmax(S[h,n,:])[m] * Vv[h,n,m]   (one wave32 per row)
__global__ __launch_bounds__(256)
void softmax_weighted_sum(const float* __restrict__ S,
                          const float* __restrict__ Vv,
                          float* __restrict__ out)
{
  const int lane = threadIdx.x & 31;
  const int row  = blockIdx.x * 8 + (threadIdx.x >> 5);   // 0 .. H*N-1
  const float* Srow = S  + (size_t)row * 256;
  const float* Vrow = Vv + (size_t)row * 256;

  float vals[8];
  float mx = -3.402823466e+38f;
  #pragma unroll
  for (int i = 0; i < 8; ++i) {
    vals[i] = Srow[lane + i * 32];
    mx = fmaxf(mx, vals[i]);
  }
  #pragma unroll
  for (int off = 16; off > 0; off >>= 1)
    mx = fmaxf(mx, __shfl_xor(mx, off, 32));

  float sum = 0.0f, dot = 0.0f;
  #pragma unroll
  for (int i = 0; i < 8; ++i) {
    const float e = __expf(vals[i] - mx);
    sum += e;
    dot += e * Vrow[lane + i * 32];
  }
  #pragma unroll
  for (int off = 16; off > 0; off >>= 1) {
    sum += __shfl_xor(sum, off, 32);
    dot += __shfl_xor(dot, off, 32);
  }
  if (lane == 0) out[row] = dot / sum;
}

extern "C" void kernel_launch(void* const* d_in, const int* in_sizes, int n_in,
                              void* d_out, int out_size, void* d_ws, size_t ws_size,
                              hipStream_t stream)
{
  (void)in_sizes; (void)n_in; (void)out_size; (void)ws_size;

  const float* x_k  = (const float*)d_in[0];
  const float* x_q  = (const float*)d_in[1];
  const float* x_v  = (const float*)d_in[2];
  const float* W_Kp = (const float*)d_in[3];
  const float* b_Kp = (const float*)d_in[4];
  const float* W_Qp = (const float*)d_in[5];
  const float* b_Qp = (const float*)d_in[6];
  const float* W_Vp = (const float*)d_in[7];
  const float* b_Vp = (const float*)d_in[8];
  const float* WK   = (const float*)d_in[9];
  const float* WQ   = (const float*)d_in[10];
  const float* WV   = (const float*)d_in[11];
  float* out = (float*)d_out;

  const int N = 256, L = 8192, H = 32, D = 256;
  const size_t SZ = (size_t)N * L;   // 2,097,152 floats == H*N*D too

  float* ws = (float*)d_ws;
  float* k  = ws + 0 * SZ;
  float* q  = ws + 1 * SZ;
  float* v  = ws + 2 * SZ;
  float* Kk = ws + 3 * SZ;
  float* Qq = ws + 4 * SZ;
  float* Vv = ws + 5 * SZ;
  float* S  = ws + 6 * SZ;

  dim3 blk(THREADS);

  // 1) pre-projections: [N,L] = [N,L] @ [L,L] + b      (128 WGs each)
  dim3 gpre(L / BN, 1, 1);
  gemm_bf16_wmma<false><<<gpre, blk, 0, stream>>>(x_k, L, 0, W_Kp, L, 0, b_Kp, k, L, 0, L);
  gemm_bf16_wmma<false><<<gpre, blk, 0, stream>>>(x_q, L, 0, W_Qp, L, 0, b_Qp, q, L, 0, L);
  gemm_bf16_wmma<false><<<gpre, blk, 0, stream>>>(x_v, L, 0, W_Vp, L, 0, b_Vp, v, L, 0, L);

  // 2) per-head projections: Kk[h] = k @ WK[h]  ([N,L]@[L,D], heads on grid.z)
  dim3 ghead(D / BN, 1, H);
  gemm_bf16_wmma<false><<<ghead, blk, 0, stream>>>(k, L, 0, WK, D, (size_t)L * D, nullptr, Kk, D, (size_t)N * D, L);
  gemm_bf16_wmma<false><<<ghead, blk, 0, stream>>>(q, L, 0, WQ, D, (size_t)L * D, nullptr, Qq, D, (size_t)N * D, L);
  gemm_bf16_wmma<false><<<ghead, blk, 0, stream>>>(v, L, 0, WV, D, (size_t)L * D, nullptr, Vv, D, (size_t)N * D, L);

  // 3) scores: S[h] = Kk[h] @ Qq[h]^T  ([N,D]@[D,N] via TRANSB)
  dim3 gsc(N / BN, 1, H);
  gemm_bf16_wmma<true><<<gsc, blk, 0, stream>>>(Kk, D, (size_t)N * D, Qq, D, (size_t)N * D, nullptr, S, N, (size_t)N * N, D);

  // 4) row softmax + elementwise weighted sum with Vv -> out[H,N]
  softmax_weighted_sum<<<(H * N) / 8, 256, 0, stream>>>(S, Vv, out);
}
